// SpatialConvOrderK_13408887898721
// MI455X (gfx1250) — matrix operand order repaired and checked by reading, not compiled, not run on hardware
//
#include <hip/hip_runtime.h>
#include <hip/hip_bf16.h>

typedef __attribute__((ext_vector_type(16))) __bf16 v16bf;
typedef __attribute__((ext_vector_type(8)))  float  v8f;

#define B_    16
#define C_    64
#define N_    1000
#define L_    24
#define SUP   3
#define NP    1024                 // node dim padded to 64 WMMA tiles
#define M_    (B_*C_*L_)           // 24576 diffusion columns (b,c,l)
#define S_    (B_*N_*L_)           // 384000 spatial sites (b,n,l)
#define KZ    256                  // effective channels in final GEMM
#define C_OUT 64

// ---------------- pack kernels ----------------

// Sb[i][w][v] bf16, zero padded to 1024x1024, row-major (K=v contiguous)
__global__ void k_pack_sup(const float* __restrict__ sup, __bf16* __restrict__ Sb) {
    size_t idx = (size_t)blockIdx.x * 256 + threadIdx.x;   // SUP*NP*NP total
    int v = idx & (NP - 1);
    int r = (idx >> 10) & (NP - 1);
    int i = idx >> 20;
    float val = 0.f;
    if (r < N_ && v < N_) val = sup[(size_t)i * N_ * N_ + (size_t)r * N_ + v];
    Sb[idx] = (__bf16)val;
}

// Xt[m][v] bf16 = x[b,c,v,l] with m=(b*C+c)*L+l ; K=v contiguous per row
__global__ void k_pack_xT(const float* __restrict__ x, __bf16* __restrict__ Xt) {
    size_t idx = (size_t)blockIdx.x * 256 + threadIdx.x;   // M_*NP total
    int v = idx & (NP - 1);
    int m = (int)(idx >> 10);
    float val = 0.f;
    if (v < N_) {
        int bc = m / L_, l = m % L_;
        val = x[((size_t)bc * N_ + v) * L_ + l];
    }
    Xt[idx] = (__bf16)val;
}

// Wb[o][k] bf16, k = g*64+c: g0=self, g1=S0, g2=S1, g3=S2 (folded duplicate)
__global__ void k_pack_w(const float* __restrict__ w, __bf16* __restrict__ Wb) {
    int idx = blockIdx.x * 256 + threadIdx.x;              // 64*256 total
    int o = idx >> 8, k = idx & 255;
    int g = k >> 6, c = k & 63;
    float v;
    if (g < 3) v = w[o * 448 + g * 64 + c];
    else       v = w[o * 448 + 192 + c] + w[o * 448 + 256 + c];
    Wb[idx] = (__bf16)v;
}

// ---------------- diffusion GEMM: Y_i = S_i * X  (bf16 WMMA, f32 acc) -------
// grid: (M_/128, NP/128, SUP), block 256 (8 waves, 2x4 wave grid, 64x32/wave).
// Double-buffered LDS fed by GLOBAL_LOAD_ASYNC_TO_LDS_B128 (ASYNCcnt-tracked):
// stage k+1's fetch overlaps stage k's 8 WMMAs; s_wait_asynccnt 4 retires the
// current stage while the next stage's 4 per-wave async loads stay in flight.
__global__ __launch_bounds__(256) void k_diffusion(const __bf16* __restrict__ Sb,
                                                   const __bf16* __restrict__ Xt,
                                                   __bf16* __restrict__ Y) {
    __shared__ __align__(16) __bf16 As[2][128 * 32];   // [buf][n_local][k] 2x8KB
    __shared__ __align__(16) __bf16 Bs[2][128 * 32];   // [buf][m_local][k] 2x8KB

    const int sup = blockIdx.z;
    const int n0  = blockIdx.y * 128;
    const int m0  = blockIdx.x * 128;
    const __bf16* Sbase = Sb + (size_t)sup * NP * NP + (size_t)n0 * NP;
    const __bf16* Xbase = Xt + (size_t)m0 * NP;
    __bf16*       Ybase = Y  + (size_t)sup * NP * M_;

    const int t    = threadIdx.x;
    const int lane = t & 31, wv = t >> 5;
    const int wn = (wv >> 2) * 64;     // wave row offset: 0 / 64
    const int wm = (wv & 3)  * 32;     // wave col offset: 0/32/64/96
    const int lr = lane & 15, hi = lane >> 4;

    // 16B chunks: tile = 512 chunks; this thread moves chunks c0,c1 of A and B
    const int c0 = t, c1 = t + 256;
    const unsigned ga0 = (unsigned)((c0 >> 2) * (NP * 2) + (c0 & 3) * 16); // global byte off
    const unsigned ga1 = (unsigned)((c1 >> 2) * (NP * 2) + (c1 & 3) * 16);
    const unsigned lbA = (unsigned)(size_t)&As[0][0]; // LDS byte offsets (low 32 bits
    const unsigned lbB = (unsigned)(size_t)&Bs[0][0]; // of generic pointer = LDS addr)
    const unsigned lA0 = lbA + (unsigned)c0 * 16, lA1 = lbA + (unsigned)c1 * 16;
    const unsigned lB0 = lbB + (unsigned)c0 * 16, lB1 = lbB + (unsigned)c1 * 16;

    auto issue = [&](int buf, int kb) {
        const unsigned kb2 = (unsigned)kb * 2;      // k-block byte offset
        const unsigned bo  = (unsigned)buf * (128 * 32 * 2);
        asm volatile("global_load_async_to_lds_b128 %0, %1, %2 offset:0"
                     :: "v"(lA0 + bo), "v"(ga0 + kb2), "s"(Sbase) : "memory");
        asm volatile("global_load_async_to_lds_b128 %0, %1, %2 offset:0"
                     :: "v"(lA1 + bo), "v"(ga1 + kb2), "s"(Sbase) : "memory");
        asm volatile("global_load_async_to_lds_b128 %0, %1, %2 offset:0"
                     :: "v"(lB0 + bo), "v"(ga0 + kb2), "s"(Xbase) : "memory");
        asm volatile("global_load_async_to_lds_b128 %0, %1, %2 offset:0"
                     :: "v"(lB1 + bo), "v"(ga1 + kb2), "s"(Xbase) : "memory");
    };

    v8f acc[4][2] = {};

    issue(0, 0);                                   // prologue: stage 0
    const int NSTAGE = NP / 32;                    // 32 k-blocks
    for (int s = 0; s < NSTAGE; ++s) {
        const int cur = s & 1;
        if (s + 1 < NSTAGE) {
            issue(cur ^ 1, (s + 1) * 32);          // prefetch next stage
            asm volatile("s_wait_asynccnt 0x4" ::: "memory");  // cur stage done
        } else {
            asm volatile("s_wait_asynccnt 0x0" ::: "memory");
        }
        __syncthreads();                           // all waves' cur loads visible

        // B fragments: lane lr = column, hi selects K 0..15 / 16..31 (contiguous)
        v16bf bfrag[2];
        for (int j = 0; j < 2; j++) {
            const uint4* p = (const uint4*)(&Bs[cur][0] + (size_t)(wm + j * 16 + lr) * 32 + hi * 16);
            union { uint4 u[2]; v16bf v; } u;
            u.u[0] = p[0]; u.u[1] = p[1];
            bfrag[j] = u.v;
        }
        // A fragments: lane lr = row, hi selects K 0..15 / 16..31 (contiguous)
        for (int i = 0; i < 4; i++) {
            const __bf16* ap = &As[cur][0] + (size_t)(wn + i * 16 + lr) * 32 + hi * 16;
            union { uint4 u[2]; v16bf v; } u;
            u.u[0] = *(const uint4*)(ap);
            u.u[1] = *(const uint4*)(ap + 8);
            v16bf afrag = u.v;
            for (int j = 0; j < 2; j++)
                acc[i][j] = __builtin_amdgcn_wmma_f32_16x16x32_bf16(
                    false, afrag, false, bfrag[j], (short)0, acc[i][j], false, false);
        }
        __syncthreads();                           // before buf cur is refilled
    }

    // C/D layout: vgpr r -> row M=r (lanes 0-15) / r+8 (lanes 16-31), col = lane%16
    for (int i = 0; i < 4; i++)
        for (int j = 0; j < 2; j++) {
            int col  = m0 + wm + j * 16 + lr;
            int rowb = n0 + wn + i * 16 + hi * 8;
            for (int r = 0; r < 8; r++)
                Ybase[(size_t)(rowb + r) * M_ + col] = (__bf16)acc[i][j][r];
        }
}

// ---------------- repack: Z[(b,n,l)][256] = {x, Y0, Y1, Y2} channels ---------
// one block per (b,n): coalesced reads, contiguous 512B row writes
__global__ __launch_bounds__(256) void k_repack(const float* __restrict__ x,
                                                const __bf16* __restrict__ Y,
                                                __bf16* __restrict__ Z) {
    __shared__ __bf16 sx[L_ * C_];            // [l][c]
    __shared__ __bf16 sy[SUP][L_ * C_];       // [l][c]
    const int bn = blockIdx.x;
    const int b = bn / N_, n = bn % N_;
    const int t = threadIdx.x;

    const float* xb = x + (size_t)b * C_ * N_ * L_ + (size_t)n * L_;
    for (int e = t; e < C_ * L_; e += 256) {
        int c = e / L_, l = e % L_;
        sx[l * C_ + c] = (__bf16)xb[(size_t)c * (N_ * L_) + l];
    }
    for (int i = 0; i < SUP; i++) {
        const __bf16* yb = Y + (size_t)i * NP * M_ + (size_t)n * M_ + (size_t)b * C_ * L_;
        for (int e = t; e < C_ * L_; e += 256) {
            int c = e / L_, l = e % L_;
            sy[i][l * C_ + c] = yb[e];
        }
    }
    __syncthreads();

    __bf16* zb = Z + (size_t)bn * L_ * KZ;
    for (int e = t; e < L_ * KZ; e += 256) {
        int l = e / KZ, k = e % KZ;
        int g = k >> 6, c = k & 63;
        zb[e] = (g == 0) ? sx[l * C_ + c] : sy[g - 1][l * C_ + c];
    }
}

// ---------------- final 1x1 conv GEMM: out = Weff * Z + bias (bf16 WMMA) -----
// grid: S_/16/8 blocks; each wave owns one 16-col s-tile and all 4 o-tiles
__global__ __launch_bounds__(256) void k_final(const __bf16* __restrict__ Wb,
                                               const __bf16* __restrict__ Z,
                                               const float* __restrict__ bias,
                                               float* __restrict__ out) {
    const int t = threadIdx.x, lane = t & 31, wv = t >> 5;
    const int lr = lane & 15, hi = lane >> 4;
    const int s0 = (blockIdx.x * 8 + wv) * 16;

    v8f acc[4] = {};
    const __bf16* zrow = Z + (size_t)(s0 + lr) * KZ + hi * 16;

    for (int ks = 0; ks < 8; ks++) {
        union { uint4 u[2]; v16bf v; } ub;
        const uint4* zp = (const uint4*)(zrow + ks * 32);
        ub.u[0] = zp[0]; ub.u[1] = zp[1];
        v16bf bf = ub.v;
        for (int ot = 0; ot < 4; ot++) {
            const __bf16* ap = Wb + (size_t)(ot * 16 + lr) * KZ + ks * 32 + hi * 16;
            union { uint4 u[2]; v16bf v; } ua;
            ua.u[0] = *(const uint4*)(ap);
            ua.u[1] = *(const uint4*)(ap + 8);
            acc[ot] = __builtin_amdgcn_wmma_f32_16x16x32_bf16(
                false, ua.v, false, bf, (short)0, acc[ot], false, false);
        }
    }

    const int s   = s0 + lr;
    const int b   = s / (N_ * L_);
    const int rem = s % (N_ * L_);
    for (int ot = 0; ot < 4; ot++)
        for (int r = 0; r < 8; r++) {
            int o = ot * 16 + hi * 8 + r;
            out[((size_t)b * C_OUT + o) * (N_ * L_) + rem] = acc[ot][r] + bias[o];
        }
}

// ---------------- launcher ----------------
extern "C" void kernel_launch(void* const* d_in, const int* in_sizes, int n_in,
                              void* d_out, int out_size, void* d_ws, size_t ws_size,
                              hipStream_t stream) {
    const float* x    = (const float*)d_in[0];
    const float* sup  = (const float*)d_in[1];
    const float* w    = (const float*)d_in[2];
    const float* bias = (const float*)d_in[3];
    float* out = (float*)d_out;

    char* ws = (char*)d_ws;
    size_t off = 0;
    __bf16* Sb = (__bf16*)(ws + off); off += (size_t)SUP * NP * NP * 2;   // 6.3 MB
    __bf16* Xt = (__bf16*)(ws + off); off += (size_t)M_ * NP * 2;         // 50.3 MB
    __bf16* Wb = (__bf16*)(ws + off); off += (size_t)C_OUT * KZ * 2;      // 32 KB
    __bf16* Y  = (__bf16*)(ws + off); off += (size_t)SUP * NP * M_ * 2;   // 151 MB
    __bf16* Z  = (__bf16*)(ws + off); off += (size_t)S_ * KZ * 2;         // 197 MB

    k_pack_sup<<<(SUP * NP * NP) / 256, 256, 0, stream>>>(sup, Sb);
    k_pack_xT<<<((size_t)M_ * NP) / 256, 256, 0, stream>>>(x, Xt);
    k_pack_w<<<(C_OUT * KZ) / 256, 256, 0, stream>>>(w, Wb);

    dim3 gd(M_ / 128, NP / 128, SUP);
    k_diffusion<<<gd, 256, 0, stream>>>(Sb, Xt, Y);

    k_repack<<<B_ * N_, 256, 0, stream>>>(x, Y, Z);

    k_final<<<(S_ / 16) / 8, 256, 0, stream>>>(Wb, Z, bias, out);
}
